// MusicGNN_86723979641477
// MI455X (gfx1250) — matrix operand order repaired and checked by reading, not compiled, not run on hardware
//
#include <hip/hip_runtime.h>

typedef __attribute__((ext_vector_type(16))) _Float16 v16h;
typedef __attribute__((ext_vector_type(8)))  float    v8f;

#define NA   100000
#define NT   300000
#define NG   2000
#define EPT  600000
#define ETG  400000
#define HID  64
#define HEADS 4
#define GDIM (HEADS*HID)   // 256

// ---------------- helpers ----------------

__device__ inline unsigned fkey(float f) {           // order-preserving float->uint
    unsigned u = __float_as_uint(f);
    return (u >> 31) ? ~u : (u | 0x80000000u);
}
__device__ inline float mdec(unsigned k) {           // decode; non-finite -> 0 (ref semantics)
    unsigned u = (k & 0x80000000u) ? (k & 0x7fffffffu) : ~k;
    float f = __uint_as_float(u);
    return __builtin_isfinite(f) ? f : 0.0f;
}

// ---------------- generic WMMA GEMM:  C[M,N] = A[M,K] @ BT[N,K]^T + bias ----------------
// A row-major (f32 or f16), BT = pre-transposed B (f16, [N][K]).
// One wave computes a 16x64 tile (4 accumulators sharing one A fragment) via
// v_wmma_f32_16x16x32_f16.  M%16==0, K%32==0, N%64==0.
template <bool AF32>
__global__ void __launch_bounds__(128)
wmma_gemm_kernel(const void* __restrict__ A_, const _Float16* __restrict__ BT,
                 const float* __restrict__ bias,
                 float* __restrict__ Cf, _Float16* __restrict__ Ch,
                 int M, int N, int K)
{
    const int wave = threadIdx.x >> 5;
    const int lane = threadIdx.x & 31;
    const int row_base = blockIdx.x * 64 + wave * 16;
    if (row_base >= M) return;
    const int col_base = blockIdx.y * 64;
    const int r  = lane & 15;
    const int hi = lane >> 4;            // half-wave select
    const size_t arow = (size_t)(row_base + r);
    const int kA = hi * 8;               // A frag: K = kA..kA+7 and 16+kA..16+kA+7
    const int kB = hi * 8;               // B frag: K = kB..kB+7 and 16+kB..16+kB+7 at column r

    v8f acc[4] = {{}, {}, {}, {}};
    for (int k0 = 0; k0 < K; k0 += 32) {
        v16h a;
        if (AF32) {
            const float* ap = (const float*)A_ + arow * K + k0 + kA;
            #pragma unroll
            for (int j = 0; j < 8; ++j) { a[j] = (_Float16)ap[j]; a[8 + j] = (_Float16)ap[16 + j]; }
        } else {
            const _Float16* ap = (const _Float16*)A_ + arow * K + k0 + kA;
            #pragma unroll
            for (int j = 0; j < 8; ++j) { a[j] = ap[j]; a[8 + j] = ap[16 + j]; }
        }
        #pragma unroll
        for (int ct = 0; ct < 4; ++ct) {
            v16h b;
            const _Float16* bp = BT + (size_t)(col_base + ct * 16 + r) * K + k0 + kB;
            #pragma unroll
            for (int j = 0; j < 8; ++j) { b[j] = bp[j]; b[8 + j] = bp[16 + j]; }
            acc[ct] = __builtin_amdgcn_wmma_f32_16x16x32_f16(false, a, false, b,
                                                             (short)0, acc[ct], false, false);
        }
    }
    #pragma unroll
    for (int ct = 0; ct < 4; ++ct) {
        const int oc = col_base + ct * 16 + r;
        const float bv = bias ? bias[oc] : 0.0f;
        #pragma unroll
        for (int v = 0; v < 8; ++v) {
            const size_t orow = (size_t)(row_base + v + hi * 8);
            float val = acc[ct][v] + bv;
            if (Cf) Cf[orow * (size_t)N + oc] = val;
            if (Ch) Ch[orow * (size_t)N + oc] = (_Float16)val;
        }
    }
}

// ---------------- weight prep: f32 [K][N] -> f16 [N][K] ----------------
__global__ void transpose_to_f16(const float* __restrict__ W, _Float16* __restrict__ BT,
                                 int K, int N)
{
    int idx = blockIdx.x * blockDim.x + threadIdx.x;
    if (idx >= K * N) return;
    int k = idx / N, n = idx % N;
    BT[(size_t)n * K + k] = (_Float16)W[idx];
}

__global__ void fill_zero_u32(unsigned* __restrict__ p, size_t n)
{
    size_t i = (size_t)blockIdx.x * blockDim.x + threadIdx.x;
    size_t stride = (size_t)gridDim.x * blockDim.x;
    for (; i < n; i += stride) p[i] = 0u;
}

// ---------------- edge pass 1: logits + segment max ----------------
// one wave per edge; lane covers 8 channels; head = lane/8
__global__ void __launch_bounds__(256)
edge_score_kernel(const float* __restrict__ gl, const float* __restrict__ gr,
                  const float* __restrict__ att,
                  const int* __restrict__ src, const int* __restrict__ dst,
                  float* __restrict__ s, unsigned* __restrict__ m, int E)
{
    int e = blockIdx.x * 8 + (threadIdx.x >> 5);
    if (e >= E) return;
    int lane = threadIdx.x & 31;
    int si = src[e], di = dst[e];
    const float* glp = gl + (size_t)si * GDIM + lane * 8;
    const float* grp = gr + (size_t)di * GDIM + lane * 8;
    const float* ap  = att + lane * 8;
    float part = 0.0f;
    #pragma unroll
    for (int j = 0; j < 8; ++j) {
        float v = glp[j] + grp[j];
        v = v > 0.0f ? v : 0.2f * v;              // leaky_relu(0.2)
        part += ap[j] * v;
    }
    part += __shfl_down(part, 4, 8);
    part += __shfl_down(part, 2, 8);
    part += __shfl_down(part, 1, 8);
    if ((lane & 7) == 0) {
        int h = lane >> 3;
        s[(size_t)e * HEADS + h] = part;
        atomicMax(m + (size_t)di * HEADS + h, fkey(part));
    }
}

// ---------------- edge pass 2: exp + segment sum ----------------
__global__ void edge_softmax_kernel(const int* __restrict__ dst, float* __restrict__ s,
                                    const unsigned* __restrict__ m, float* __restrict__ den,
                                    int E)
{
    int idx = blockIdx.x * blockDim.x + threadIdx.x;
    if (idx >= E * HEADS) return;
    int e = idx >> 2, h = idx & 3;
    int di = dst[e];
    float ex = expf(s[idx] - mdec(m[(size_t)di * HEADS + h]));
    s[idx] = ex;
    atomicAdd(den + (size_t)di * HEADS + h, ex);
}

// ---------------- edge pass 3: weighted scatter-add ----------------
__global__ void __launch_bounds__(256)
edge_aggregate_kernel(const float* __restrict__ gl, const float* __restrict__ ex,
                      const float* __restrict__ den,
                      const int* __restrict__ src, const int* __restrict__ dst,
                      float* __restrict__ O, int E)
{
    int e = blockIdx.x * 8 + (threadIdx.x >> 5);
    if (e >= E) return;
    int lane = threadIdx.x & 31;
    int h = lane >> 3;
    int si = src[e], di = dst[e];
    float alpha = ex[(size_t)e * HEADS + h] / (den[(size_t)di * HEADS + h] + 1e-16f);
    const float* glp = gl + (size_t)si * GDIM + lane * 8;
    float* op = O + (size_t)di * GDIM + lane * 8;
    #pragma unroll
    for (int j = 0; j < 8; ++j) atomicAdd(op + j, alpha * glp[j]);
}

// ---------------- head mean + bias + ELU -> f16 for next WMMA GEMM ----------------
__global__ void finalize_elu_kernel(const float* __restrict__ O, const float* __restrict__ bias,
                                    _Float16* __restrict__ Hh, int Nn)
{
    int idx = blockIdx.x * blockDim.x + threadIdx.x;
    if (idx >= Nn * HID) return;
    int n = idx >> 6, c = idx & 63;
    const float* p = O + (size_t)n * GDIM + c;
    float v = 0.25f * (p[0] + p[64] + p[128] + p[192]) + bias[c];
    v = v > 0.0f ? v : expm1f(v);                 // elu
    Hh[idx] = (_Float16)v;
}

// ---------------- in-place row L2 norm (64 cols), one wave per row ----------------
__global__ void __launch_bounds__(256)
l2norm_kernel(float* __restrict__ y, int rows)
{
    int row = blockIdx.x * 8 + (threadIdx.x >> 5);
    if (row >= rows) return;
    int lane = threadIdx.x & 31;
    float* p = y + (size_t)row * 64 + lane * 2;
    float a = p[0], b = p[1];
    float ss = a * a + b * b;
    #pragma unroll
    for (int o = 16; o; o >>= 1) ss += __shfl_xor(ss, o);
    float inv = 1.0f / fmaxf(sqrtf(ss), 1e-12f);
    p[0] = a * inv; p[1] = b * inv;
}

// ---------------- orchestration ----------------
extern "C" void kernel_launch(void* const* d_in, const int* in_sizes, int n_in,
                              void* d_out, int out_size, void* d_ws, size_t ws_size,
                              hipStream_t stream)
{
    (void)in_sizes; (void)n_in; (void)out_size; (void)ws_size;
    const float* x_a    = (const float*)d_in[0];
    const float* x_t    = (const float*)d_in[1];
    const float* x_g    = (const float*)d_in[2];
    const float* enc_Wa = (const float*)d_in[3];
    const float* enc_ba = (const float*)d_in[4];
    const float* enc_Wt = (const float*)d_in[5];
    const float* enc_bt = (const float*)d_in[6];
    const float* enc_Wg = (const float*)d_in[7];
    const float* enc_bg = (const float*)d_in[8];
    const int*   src_pt = (const int*)d_in[9];
    const int*   dst_pt = (const int*)d_in[10];
    const int*   src_tg = (const int*)d_in[11];
    const int*   dst_tg = (const int*)d_in[12];
    // only layer l=1 matters (reference loop overwrites h each iteration)
    const int WOFF = 1 * HID * HEADS * HID;   // 16384
    const int BOFF = 1 * HEADS * HID;         // 256
    const int AOFF = 1 * HEADS * HID;         // 256
    const int NOFF = 1 * HID;                 // 64
    const float* Wl_pt  = (const float*)d_in[13] + WOFF;
    const float* bl_pt  = (const float*)d_in[14] + BOFF;
    const float* Wr_pt  = (const float*)d_in[15] + WOFF;
    const float* br_pt  = (const float*)d_in[16] + BOFF;
    const float* att_pt = (const float*)d_in[17] + AOFF;
    const float* bias_pt= (const float*)d_in[18] + NOFF;
    const float* Wl_tg  = (const float*)d_in[19] + WOFF;
    const float* bl_tg  = (const float*)d_in[20] + BOFF;
    const float* Wr_tg  = (const float*)d_in[21] + WOFF;
    const float* br_tg  = (const float*)d_in[22] + BOFF;
    const float* att_tg = (const float*)d_in[23] + AOFF;
    const float* bias_tg= (const float*)d_in[24] + NOFF;
    const float* lin_tW = (const float*)d_in[25];
    const float* lin_tb = (const float*)d_in[26];
    const float* lin_gW = (const float*)d_in[27];
    const float* lin_gb = (const float*)d_in[28];

    // workspace carve-out
    char* wsb = (char*)d_ws;
    size_t off = 0;
    auto alloc = [&](size_t bytes) -> void* {
        off = (off + 255) & ~(size_t)255;
        void* p = wsb + off; off += bytes; return p;
    };
    _Float16* bt_enc_a = (_Float16*)alloc((size_t)64 * 64 * 2);
    _Float16* bt_enc_t = (_Float16*)alloc((size_t)64 * 96 * 2);
    _Float16* bt_enc_g = (_Float16*)alloc((size_t)64 * 32 * 2);
    _Float16* bt_Wl_pt = (_Float16*)alloc((size_t)GDIM * 64 * 2);
    _Float16* bt_Wr_pt = (_Float16*)alloc((size_t)GDIM * 64 * 2);
    _Float16* bt_Wl_tg = (_Float16*)alloc((size_t)GDIM * 64 * 2);
    _Float16* bt_Wr_tg = (_Float16*)alloc((size_t)GDIM * 64 * 2);
    _Float16* bt_lin_t = (_Float16*)alloc((size_t)64 * 64 * 2);
    _Float16* bt_lin_g = (_Float16*)alloc((size_t)64 * 64 * 2);
    _Float16* enc_a_h  = (_Float16*)alloc((size_t)NA * HID * 2);
    _Float16* enc_t_h  = (_Float16*)alloc((size_t)NT * HID * 2);
    _Float16* enc_g_h  = (_Float16*)alloc((size_t)NG * HID * 2);
    float*    GL       = (float*)alloc((size_t)NT * GDIM * 4);  // gl_pt, later gl_tg
    float*    GR       = (float*)alloc((size_t)NT * GDIM * 4);  // gr_pt -> O_pt; later gr_tg/O_tg
    float*    sbuf     = (float*)alloc((size_t)EPT * HEADS * 4);
    unsigned* mbuf     = (unsigned*)alloc((size_t)NT * HEADS * 4);
    float*    dbuf     = (float*)alloc((size_t)NT * HEADS * 4);
    _Float16* h_t      = (_Float16*)alloc((size_t)NT * HID * 2);
    _Float16* h_g      = (_Float16*)alloc((size_t)NG * HID * 2);

    auto cdiv = [](int a, int b) { return (a + b - 1) / b; };

    // --- weight prep (transpose + f16) ---
    transpose_to_f16<<<cdiv(64 * 64, 256), 256, 0, stream>>>(enc_Wa, bt_enc_a, 64, 64);
    transpose_to_f16<<<cdiv(96 * 64, 256), 256, 0, stream>>>(enc_Wt, bt_enc_t, 96, 64);
    transpose_to_f16<<<cdiv(32 * 64, 256), 256, 0, stream>>>(enc_Wg, bt_enc_g, 32, 64);
    transpose_to_f16<<<cdiv(64 * GDIM, 256), 256, 0, stream>>>(Wl_pt, bt_Wl_pt, 64, GDIM);
    transpose_to_f16<<<cdiv(64 * GDIM, 256), 256, 0, stream>>>(Wr_pt, bt_Wr_pt, 64, GDIM);
    transpose_to_f16<<<cdiv(64 * GDIM, 256), 256, 0, stream>>>(Wl_tg, bt_Wl_tg, 64, GDIM);
    transpose_to_f16<<<cdiv(64 * GDIM, 256), 256, 0, stream>>>(Wr_tg, bt_Wr_tg, 64, GDIM);
    transpose_to_f16<<<cdiv(64 * 64, 256), 256, 0, stream>>>(lin_tW, bt_lin_t, 64, 64);
    transpose_to_f16<<<cdiv(64 * 64, 256), 256, 0, stream>>>(lin_gW, bt_lin_g, 64, 64);

    // --- encoders (WMMA, f32 A converted on the fly, f16 output) ---
    wmma_gemm_kernel<true><<<dim3(cdiv(NA, 64), 1), 128, 0, stream>>>(
        x_a, bt_enc_a, enc_ba, nullptr, enc_a_h, NA, 64, 64);
    wmma_gemm_kernel<true><<<dim3(cdiv(NT, 64), 1), 128, 0, stream>>>(
        x_t, bt_enc_t, enc_bt, nullptr, enc_t_h, NT, 64, 96);
    wmma_gemm_kernel<true><<<dim3(cdiv(NG, 64), 1), 128, 0, stream>>>(
        x_g, bt_enc_g, enc_bg, nullptr, enc_g_h, NG, 64, 32);

    // ================= PT edge type: artist -> track =================
    wmma_gemm_kernel<false><<<dim3(cdiv(NA, 64), GDIM / 64), 128, 0, stream>>>(
        enc_a_h, bt_Wl_pt, bl_pt, GL, nullptr, NA, GDIM, 64);
    wmma_gemm_kernel<false><<<dim3(cdiv(NT, 64), GDIM / 64), 128, 0, stream>>>(
        enc_t_h, bt_Wr_pt, br_pt, GR, nullptr, NT, GDIM, 64);
    fill_zero_u32<<<2048, 256, 0, stream>>>(mbuf, (size_t)NT * HEADS);
    fill_zero_u32<<<2048, 256, 0, stream>>>((unsigned*)dbuf, (size_t)NT * HEADS);
    edge_score_kernel<<<cdiv(EPT, 8), 256, 0, stream>>>(GL, GR, att_pt, src_pt, dst_pt,
                                                        sbuf, mbuf, EPT);
    edge_softmax_kernel<<<cdiv(EPT * HEADS, 256), 256, 0, stream>>>(dst_pt, sbuf, mbuf, dbuf, EPT);
    fill_zero_u32<<<4096, 256, 0, stream>>>((unsigned*)GR, (size_t)NT * GDIM);   // GR becomes O_pt
    edge_aggregate_kernel<<<cdiv(EPT, 8), 256, 0, stream>>>(GL, sbuf, dbuf, src_pt, dst_pt,
                                                            GR, EPT);
    finalize_elu_kernel<<<cdiv(NT * HID, 256), 256, 0, stream>>>(GR, bias_pt, h_t, NT);

    // ================= TG edge type: track -> genre =================
    wmma_gemm_kernel<false><<<dim3(cdiv(NT, 64), GDIM / 64), 128, 0, stream>>>(
        enc_t_h, bt_Wl_tg, bl_tg, GL, nullptr, NT, GDIM, 64);
    wmma_gemm_kernel<false><<<dim3(cdiv(NG, 64), GDIM / 64), 128, 0, stream>>>(
        enc_g_h, bt_Wr_tg, br_tg, GR, nullptr, NG, GDIM, 64);              // gr_tg
    fill_zero_u32<<<64, 256, 0, stream>>>(mbuf, (size_t)NG * HEADS);
    fill_zero_u32<<<64, 256, 0, stream>>>((unsigned*)dbuf, (size_t)NG * HEADS);
    edge_score_kernel<<<cdiv(ETG, 8), 256, 0, stream>>>(GL, GR, att_tg, src_tg, dst_tg,
                                                        sbuf, mbuf, ETG);
    edge_softmax_kernel<<<cdiv(ETG * HEADS, 256), 256, 0, stream>>>(dst_tg, sbuf, mbuf, dbuf, ETG);
    float* Otg = GR + (size_t)NG * GDIM;
    fill_zero_u32<<<512, 256, 0, stream>>>((unsigned*)Otg, (size_t)NG * GDIM);
    edge_aggregate_kernel<<<cdiv(ETG, 8), 256, 0, stream>>>(GL, sbuf, dbuf, src_tg, dst_tg,
                                                            Otg, ETG);
    finalize_elu_kernel<<<cdiv(NG * HID, 256), 256, 0, stream>>>(Otg, bias_tg, h_g, NG);

    // ================= output heads: linear (WMMA) + L2 norm (in place in d_out) =========
    float* out_track = (float*)d_out;
    float* out_genre = out_track + (size_t)NT * 64;
    wmma_gemm_kernel<false><<<dim3(cdiv(NT, 64), 1), 128, 0, stream>>>(
        h_t, bt_lin_t, lin_tb, out_track, nullptr, NT, 64, 64);
    wmma_gemm_kernel<false><<<dim3(cdiv(NG, 64), 1), 128, 0, stream>>>(
        h_g, bt_lin_g, lin_gb, out_genre, nullptr, NG, 64, 64);
    l2norm_kernel<<<cdiv(NT, 8), 256, 0, stream>>>(out_track, NT);
    l2norm_kernel<<<cdiv(NG, 8), 256, 0, stream>>>(out_genre, NG);
}
